// Encoder_46600395162446
// MI455X (gfx1250) — compile-verified, matrix operand order checked by reading, not run on hardware
//
#include <hip/hip_runtime.h>
#include <hip/hip_bf16.h>
#include <stdint.h>

typedef unsigned short u16;
typedef __attribute__((ext_vector_type(16))) __bf16 v16bf;
typedef __attribute__((ext_vector_type(8)))  float  v8f;

#define MEMR 2000
#define MEMP 2016
#define SHRINK_THRES 0.00025f
static const long COL_ELEMS = 33554432L; // 64MB bf16 im2col tile

static inline int cdiv(long a, long b){ return (int)((a + b - 1) / b); }

__device__ __forceinline__ u16 f2bf(float f){
  union { float f; uint32_t u; } v; v.f = f;
  uint32_t u = v.u;
  uint32_t r = (u + 0x7FFFu + ((u >> 16) & 1u)) >> 16;
  return (u16)r;
}

union FragBF { v16bf v; uint32_t u[8]; };

// Load one 16x32 bf16 fragment (CDNA5 A/B layout) from a K-contiguous row block.
__device__ __forceinline__ void load_frag(FragBF& f, const u16* __restrict__ p,
                                          int kb, int hsel){
#pragma unroll
  for (int r = 0; r < 8; ++r){
    int k = kb + ((r >> 2) << 4) + hsel * 8 + (r & 3) * 2;
    f.u[r] = *(const uint32_t*)(p + k);
  }
}

// ---------------- GEMM kernels (2x2 register-blocked, 32x32 per wave) -------
// A: [M,K] row-major bf16 (M multiple of 32, K multiple of 32)
// Bt: [Nn,K] row-major bf16 (row n = column n of B); Nn multiple of 16.
// Each wave computes rows [tm*32, tm*32+32) x cols [tn*32, tn*32+32).
// When Nn/16 is odd, the last wave column runs a specialized 32x16 loop
// (n2ok is wave-uniform, so EXEC stays all-ones around every WMMA).

struct Acc4 { v8f a00, a10, a01, a11; };

__device__ __forceinline__ Acc4 wmma_kloop2(const u16* __restrict__ A, int lda,
                                            const u16* __restrict__ B, int ldb,
                                            int K, int lane, bool n2ok){
  Acc4 ac;
  ac.a00 = v8f{0.f,0.f,0.f,0.f,0.f,0.f,0.f,0.f};
  ac.a10 = ac.a00; ac.a01 = ac.a00; ac.a11 = ac.a00;
  int r16  = lane & 15;
  int hsel = (lane >> 4) & 1;
  const u16* a0 = A + (long)r16 * lda;
  const u16* a1 = a0 + (long)16 * lda;
  const u16* b0 = B + (long)r16 * ldb;
  const u16* b1 = b0 + (long)16 * ldb;
  if (n2ok){
    // hot path: branch-free 2x2 body, 8 B128 loads -> 4 WMMAs per K-step
    for (int kb = 0; kb < K; kb += 32){
      FragBF fa0, fa1, fb0, fb1;
      load_frag(fa0, a0, kb, hsel);
      load_frag(fa1, a1, kb, hsel);
      load_frag(fb0, b0, kb, hsel);
      load_frag(fb1, b1, kb, hsel);
      ac.a00 = __builtin_amdgcn_wmma_f32_16x16x32_bf16(false, fa0.v, false, fb0.v,
                                                       (short)0, ac.a00, false, false);
      ac.a10 = __builtin_amdgcn_wmma_f32_16x16x32_bf16(false, fa1.v, false, fb0.v,
                                                       (short)0, ac.a10, false, false);
      ac.a01 = __builtin_amdgcn_wmma_f32_16x16x32_bf16(false, fa0.v, false, fb1.v,
                                                       (short)0, ac.a01, false, false);
      ac.a11 = __builtin_amdgcn_wmma_f32_16x16x32_bf16(false, fa1.v, false, fb1.v,
                                                       (short)0, ac.a11, false, false);
    }
  } else {
    // odd-edge path: 32x16 tile
    for (int kb = 0; kb < K; kb += 32){
      FragBF fa0, fa1, fb0;
      load_frag(fa0, a0, kb, hsel);
      load_frag(fa1, a1, kb, hsel);
      load_frag(fb0, b0, kb, hsel);
      ac.a00 = __builtin_amdgcn_wmma_f32_16x16x32_bf16(false, fa0.v, false, fb0.v,
                                                       (short)0, ac.a00, false, false);
      ac.a10 = __builtin_amdgcn_wmma_f32_16x16x32_bf16(false, fa1.v, false, fb0.v,
                                                       (short)0, ac.a10, false, false);
    }
  }
  return ac;
}

// C[M,Nn] (fp32, ldc) = A[M,K]*Bt[Nn,K]^T
__global__ void k_gemm_plain(const u16* __restrict__ A, int lda,
                             const u16* __restrict__ Bt, int ldb,
                             float* __restrict__ C, int ldc,
                             int M, int Nn, int K, int totalWaves){
  int wave = blockIdx.x * (blockDim.x >> 5) + (threadIdx.x >> 5);
  int lane = threadIdx.x & 31;
  if (wave >= totalWaves) return;
  int tilesN16 = Nn >> 4;
  int tilesN32 = (tilesN16 + 1) >> 1;
  int tm = wave / tilesN32, tn = wave % tilesN32;
  bool n2ok = (tn * 2 + 1) < tilesN16;
  Acc4 ac = wmma_kloop2(A + (long)tm * 32 * lda, lda,
                        Bt + (long)tn * 32 * ldb, ldb, K, lane, n2ok);
  int nl = lane & 15, hs = (lane >> 4) & 1;
  int m0 = tm * 32, n0 = tn * 32;
#pragma unroll
  for (int r = 0; r < 8; ++r){
    int m = r + hs * 8;
    C[(long)(m0 + m) * ldc + n0 + nl]      = ac.a00[r];
    C[(long)(m0 + 16 + m) * ldc + n0 + nl] = ac.a10[r];
  }
  if (n2ok){
#pragma unroll
    for (int r = 0; r < 8; ++r){
      int m = r + hs * 8;
      C[(long)(m0 + m) * ldc + n0 + 16 + nl]      = ac.a01[r];
      C[(long)(m0 + 16 + m) * ldc + n0 + 16 + nl] = ac.a11[r];
    }
  }
}

// Conv GEMM: A = weights[Cout,Kp], Bt = im2col[cnt,Kp]; epilogue scatters into
// NCHW dest (Ctot channels, channel offset coff) with bias (+BN)(+ReLU).
__device__ __forceinline__ void conv_store16(const v8f& acc, int m0, int n0,
                                             int lane, int p0, int HoWo,
                                             float* __restrict__ out, int Ctot, int coff,
                                             const float* __restrict__ bias,
                                             const float* __restrict__ gam,
                                             const float* __restrict__ bet,
                                             int relu){
  int nl = lane & 15, hs = (lane >> 4) & 1;
  int p  = p0 + n0 + nl;
  int nb = p / HoWo, hw = p % HoWo;
  float invs = rsqrtf(1.0f + 1e-5f);
#pragma unroll
  for (int r = 0; r < 8; ++r){
    int m = m0 + r + hs * 8;
    float v = acc[r] + bias[m];
    if (gam) v = v * gam[m] * invs + bet[m];
    if (relu) v = fmaxf(v, 0.f);
    out[((long)nb * Ctot + coff + m) * HoWo + hw] = v;
  }
}

__global__ void k_gemm_conv(const u16* __restrict__ A, int lda,
                            const u16* __restrict__ Bt, int ldb,
                            int M, int Ncnt, int K, int p0, int HoWo,
                            float* __restrict__ out, int Ctot, int coff,
                            const float* __restrict__ bias,
                            const float* __restrict__ gam,
                            const float* __restrict__ bet,
                            int relu, int totalWaves){
  int wave = blockIdx.x * (blockDim.x >> 5) + (threadIdx.x >> 5);
  int lane = threadIdx.x & 31;
  if (wave >= totalWaves) return;
  int tilesN16 = Ncnt >> 4;
  int tilesN32 = (tilesN16 + 1) >> 1;
  int tm = wave / tilesN32, tn = wave % tilesN32;
  bool n2ok = (tn * 2 + 1) < tilesN16;
  Acc4 ac = wmma_kloop2(A + (long)tm * 32 * lda, lda,
                        Bt + (long)tn * 32 * ldb, ldb, K, lane, n2ok);
  int m0 = tm * 32, n0 = tn * 32;
  conv_store16(ac.a00, m0,      n0,      lane, p0, HoWo, out, Ctot, coff, bias, gam, bet, relu);
  conv_store16(ac.a10, m0 + 16, n0,      lane, p0, HoWo, out, Ctot, coff, bias, gam, bet, relu);
  if (n2ok){
    conv_store16(ac.a01, m0,      n0 + 16, lane, p0, HoWo, out, Ctot, coff, bias, gam, bet, relu);
    conv_store16(ac.a11, m0 + 16, n0 + 16, lane, p0, HoWo, out, Ctot, coff, bias, gam, bet, relu);
  }
}

// ---------------- data movement / prep kernels ----------------

// weights fp32 -> bf16 [Cout, Kp] row-major, K = Cin*9 zero-padded to Kp.
// convT: source is w[Cin][Cout][3][3] with spatial flip.
__global__ void k_wprep(const float* __restrict__ w, int Cout, int Cin, int Kp,
                        int convT, u16* __restrict__ dst, long total){
  long t = (long)blockIdx.x * blockDim.x + threadIdx.x;
  if (t >= total) return;
  int k  = (int)(t % Kp);
  int co = (int)(t / Kp);
  u16 v = 0;
  if (k < Cin * 9){
    int kw = k % 3, kh = (k / 3) % 3, ci = k / 9;
    float f;
    if (!convT) f = w[(((long)co * Cin + ci) * 3 + kh) * 3 + kw];
    else        f = w[(((long)ci * Cout + co) * 3 + (2 - kh)) * 3 + (2 - kw)];
    v = f2bf(f);
  }
  dst[t] = v;
}

// im2col (fp32 NCHW -> bf16 [cnt, Kp]); convT path implements lhs_dilation=2,
// pad (1,2) as in the reference's transpose conv.
__global__ void k_im2col(const float* __restrict__ x, int Cin, int Hi, int Wi,
                         int Ho, int Wo, int stride, int convT,
                         int p0, int cnt, int Kp, u16* __restrict__ col, long total){
  long t = (long)blockIdx.x * blockDim.x + threadIdx.x;
  if (t >= total) return;
  int k  = (int)(t % Kp);
  int pi = (int)(t / Kp);
  u16 v = 0;
  if (k < Cin * 9){
    int kw = k % 3, kh = (k / 3) % 3, ci = k / 9;
    int p  = p0 + pi;
    int HoWo = Ho * Wo;
    int nb = p / HoWo, hw = p % HoWo, oh = hw / Wo, ow = hw % Wo;
    float f = 0.f;
    if (!convT){
      int ih = oh * stride - 1 + kh, iw = ow * stride - 1 + kw;
      if (ih >= 0 && ih < Hi && iw >= 0 && iw < Wi)
        f = x[(((long)nb * Cin + ci) * Hi + ih) * Wi + iw];
    } else {
      int dh = oh - 1 + kh, dw = ow - 1 + kw;
      if (dh >= 0 && dw >= 0 && !(dh & 1) && !(dw & 1)){
        int ih = dh >> 1, iw = dw >> 1;
        if (ih < Hi && iw < Wi)
          f = x[(((long)nb * Cin + ci) * Hi + ih) * Wi + iw];
      }
    }
    v = f2bf(f);
  }
  col[t] = v;
}

// memory matrix fp32 [2000,fd] -> bf16 [2016,fd] and transposed bf16 [fd,2016]
__global__ void k_memprep(const float* __restrict__ M, int fd,
                          u16* __restrict__ Mbf, u16* __restrict__ Mt, long total){
  long t = (long)blockIdx.x * blockDim.x + threadIdx.x;
  if (t >= total) return;
  int c = (int)(t % fd);
  int r = (int)(t / fd);
  float f = (r < MEMR) ? M[(long)r * fd + c] : 0.f;
  u16 v = f2bf(f);
  Mbf[t] = v;
  Mt[(long)c * MEMP + r] = v;
}

// _mem_window query gather: x fp32 NCHW -> q bf16 [Q, fd]
__global__ void k_gather_window(const float* __restrict__ x, u16* __restrict__ q,
                                int C, int H, int W, int wsz, int g, int fd, long total){
  long t = (long)blockIdx.x * blockDim.x + threadIdx.x;
  if (t >= total) return;
  int e = (int)(t % fd); long row = t / fd;
  int cg = C / g, Hb = H / wsz, Wb = W / wsz;
  int wb = (int)(row % Wb); row /= Wb;
  int hb = (int)(row % Hb); row /= Hb;
  int gi = (int)(row % g);  int nb = (int)(row / g);
  int ww = e % wsz, wh = (e / wsz) % wsz, cq = e / (wsz * wsz);
  q[t] = f2bf(x[(((long)nb * C + gi * cg + cq) * H + hb * wsz + wh) * W + wb * wsz + ww]);
}

__global__ void k_scatter_window(const float* __restrict__ y, float* __restrict__ dst,
                                 int C, int H, int W, int wsz, int g, int fd,
                                 int Ctot, int coff, long total){
  long t = (long)blockIdx.x * blockDim.x + threadIdx.x;
  if (t >= total) return;
  int e = (int)(t % fd); long row = t / fd;
  int cg = C / g, Hb = H / wsz, Wb = W / wsz;
  int wb = (int)(row % Wb); row /= Wb;
  int hb = (int)(row % Hb); row /= Hb;
  int gi = (int)(row % g);  int nb = (int)(row / g);
  int ww = e % wsz, wh = (e / wsz) % wsz, cq = e / (wsz * wsz);
  dst[(((long)nb * Ctot + coff + gi * cg + cq) * H + hb * wsz + wh) * W + wb * wsz + ww] = y[t];
}

// _mem_new query gather (flat order n,hb,wb,c,wh,ww), q viewed [Q,fd] row-major
__global__ void k_gather_new(const float* __restrict__ x, u16* __restrict__ q,
                             int C, int H, int W, int wsz, long total){
  long t = (long)blockIdx.x * blockDim.x + threadIdx.x;
  if (t >= total) return;
  long s = t;
  int Hb = H / wsz, Wb = W / wsz;
  int ww = (int)(s % wsz); s /= wsz;
  int wh = (int)(s % wsz); s /= wsz;
  int c  = (int)(s % C);   s /= C;
  int wb = (int)(s % Wb);  s /= Wb;
  int hb = (int)(s % Hb);  int nb = (int)(s / Hb);
  q[t] = f2bf(x[(((long)nb * C + c) * H + hb * wsz + wh) * W + wb * wsz + ww]);
}

__global__ void k_scatter_new(const float* __restrict__ y, float* __restrict__ dst,
                              int C, int H, int W, int wsz, int Ctot, int coff, long total){
  long t = (long)blockIdx.x * blockDim.x + threadIdx.x;
  if (t >= total) return;
  long s = t;
  int Hb = H / wsz, Wb = W / wsz;
  int ww = (int)(s % wsz); s /= wsz;
  int wh = (int)(s % wsz); s /= wsz;
  int c  = (int)(s % C);   s /= C;
  int wb = (int)(s % Wb);  s /= Wb;
  int hb = (int)(s % Hb);  int nb = (int)(s / Hb);
  dst[(((long)nb * Ctot + coff + c) * H + hb * wsz + wh) * W + wb * wsz + ww] = y[t];
}

// softmax over 2000 cols (stride 2016) + hard-shrink + L1 renorm -> bf16
__global__ void k_softmax(const float* __restrict__ att, u16* __restrict__ attbf){
  int row = blockIdx.x;
  int tid = threadIdx.x; // 256
  const float* a = att + (long)row * MEMP;
  u16* o = attbf + (long)row * MEMP;
  __shared__ float red[256];
  float v[8];
  float mx = -1e30f;
#pragma unroll
  for (int i = 0; i < 8; ++i){
    int c = tid + i * 256;
    float f = (c < MEMR) ? a[c] : -1e30f;
    v[i] = f; mx = fmaxf(mx, f);
  }
  red[tid] = mx; __syncthreads();
  for (int s = 128; s > 0; s >>= 1){ if (tid < s) red[tid] = fmaxf(red[tid], red[tid + s]); __syncthreads(); }
  mx = red[0]; __syncthreads();
  float sum = 0.f;
#pragma unroll
  for (int i = 0; i < 8; ++i){
    int c = tid + i * 256;
    float e = (c < MEMR) ? __expf(v[i] - mx) : 0.f;
    v[i] = e; sum += e;
  }
  red[tid] = sum; __syncthreads();
  for (int s = 128; s > 0; s >>= 1){ if (tid < s) red[tid] += red[tid + s]; __syncthreads(); }
  sum = red[0]; __syncthreads();
  float inv = 1.0f / sum;
  float sum2 = 0.f;
#pragma unroll
  for (int i = 0; i < 8; ++i){
    float p = v[i] * inv;
    float d = p - SHRINK_THRES;
    float p2 = fmaxf(d, 0.f) * p / (fabsf(d) + 1e-12f);
    v[i] = p2; sum2 += p2;
  }
  red[tid] = sum2; __syncthreads();
  for (int s = 128; s > 0; s >>= 1){ if (tid < s) red[tid] += red[tid + s]; __syncthreads(); }
  sum2 = red[0]; __syncthreads();
  float inv2 = 1.0f / (sum2 + 1e-12f);
#pragma unroll
  for (int i = 0; i < 8; ++i){
    int c = tid + i * 256;
    if (c < MEMP) o[c] = (c < MEMR) ? f2bf(v[i] * inv2) : (u16)0;
  }
}

// final 1x1 conv 64->3 at 128x128 (M=3, too small for WMMA tiles)
__global__ void k_conv1x1(const float* __restrict__ x, const float* __restrict__ w,
                          const float* __restrict__ b, float* __restrict__ out){
  int t = blockIdx.x * blockDim.x + threadIdx.x;
  if (t >= 2 * 3 * 16384) return;
  int hw = t % 16384, co = (t / 16384) % 3, nb = t / (3 * 16384);
  float s = b[co];
#pragma unroll 8
  for (int ci = 0; ci < 64; ++ci)
    s += x[((long)nb * 64 + ci) * 16384 + hw] * w[co * 64 + ci];
  out[t] = s;
}

// extract channel slice [coff,coff+Cc) of an NCHW buffer into a packed dest
__global__ void k_copych(const float* __restrict__ src, int Ctot, int coff,
                         int Cc, int HW, float* __restrict__ dst, long total){
  long t = (long)blockIdx.x * blockDim.x + threadIdx.x;
  if (t >= total) return;
  int hw = (int)(t % HW);
  int c  = (int)((t / HW) % Cc);
  int nb = (int)(t / ((long)HW * Cc));
  dst[t] = src[((long)nb * Ctot + coff + c) * HW + hw];
}

// ---------------- host-side drivers ----------------

struct Scratch {
  u16 *wbf, *col, *mbf, *mt, *qbf, *attbf;
  float *att, *ybuf;
};

static inline int gemm_waves(int M, int Nn){
  int tilesN16 = Nn >> 4;
  int tilesN32 = (tilesN16 + 1) >> 1;
  return (M >> 5) * tilesN32;
}

static void conv_layer(hipStream_t s, const Scratch& sc,
                       const float* x, int Nb, int Cin, int Hi, int Wi,
                       const float* w, const float* bias,
                       const float* gam, const float* bet,
                       int relu, int stride, int convT,
                       float* out, int Ctot, int coff, int Cout){
  int Ho = convT ? Hi * 2 : ((Hi - 1) / stride + 1);
  int Wo = convT ? Wi * 2 : ((Wi - 1) / stride + 1);
  int K = Cin * 9, Kp = (K + 31) & ~31;
  long wtot = (long)Cout * Kp;
  k_wprep<<<cdiv(wtot, 256), 256, 0, s>>>(w, Cout, Cin, Kp, convT, sc.wbf, wtot);
  int NP = Nb * Ho * Wo;
  int NPC = (int)((COL_ELEMS / (long)Kp) & ~15L);
  if (NPC > NP) NPC = NP;
  if (NPC < 16) NPC = 16;
  for (int p0 = 0; p0 < NP; p0 += NPC){
    int cnt = NP - p0; if (cnt > NPC) cnt = NPC;
    long ctot = (long)cnt * Kp;
    k_im2col<<<cdiv(ctot, 256), 256, 0, s>>>(x, Cin, Hi, Wi, Ho, Wo, stride, convT,
                                             p0, cnt, Kp, sc.col, ctot);
    int waves = gemm_waves(Cout, cnt);
    k_gemm_conv<<<cdiv(waves, 8), 256, 0, s>>>(sc.wbf, Kp, sc.col, Kp, Cout, cnt, Kp,
                                               p0, Ho * Wo, out, Ctot, coff,
                                               bias, gam, bet, relu, waves);
  }
}

static void mem_module(hipStream_t s, const Scratch& sc,
                       const float* x, int Nb, int C, int H, int W,
                       const float* M, int fd, int wsz, int g, int isWindow,
                       float* dst, int Ctot, int coff){
  int Q = isWindow ? Nb * g * (H / wsz) * (W / wsz) : (Nb * C * H * W) / fd;
  long mtot = (long)MEMP * fd;
  k_memprep<<<cdiv(mtot, 256), 256, 0, s>>>(M, fd, sc.mbf, sc.mt, mtot);
  long qt = (long)Q * fd;
  if (isWindow)
    k_gather_window<<<cdiv(qt, 256), 256, 0, s>>>(x, sc.qbf, C, H, W, wsz, g, fd, qt);
  else
    k_gather_new<<<cdiv(qt, 256), 256, 0, s>>>(x, sc.qbf, C, H, W, wsz, qt);
  int waves1 = gemm_waves(Q, MEMP);
  k_gemm_plain<<<cdiv(waves1, 8), 256, 0, s>>>(sc.qbf, fd, sc.mbf, fd,
                                               sc.att, MEMP, Q, MEMP, fd, waves1);
  k_softmax<<<Q, 256, 0, s>>>(sc.att, sc.attbf);
  int waves2 = gemm_waves(Q, fd);
  k_gemm_plain<<<cdiv(waves2, 8), 256, 0, s>>>(sc.attbf, MEMP, sc.mt, MEMP,
                                               sc.ybuf, fd, Q, fd, MEMP, waves2);
  if (isWindow)
    k_scatter_window<<<cdiv(qt, 256), 256, 0, s>>>(sc.ybuf, dst, C, H, W, wsz, g, fd, Ctot, coff, qt);
  else
    k_scatter_new<<<cdiv(qt, 256), 256, 0, s>>>(sc.ybuf, dst, C, H, W, wsz, Ctot, coff, qt);
}

struct DCp { const float *w1,*b1,*g1,*be1,*w2,*b2,*g2,*be2; };

extern "C" void kernel_launch(void* const* d_in, const int* in_sizes, int n_in,
                              void* d_out, int out_size, void* d_ws, size_t ws_size,
                              hipStream_t stream){
  (void)in_sizes; (void)out_size;
  if (n_in < 102) return;
  int idx = 0;
  auto F = [&]() -> const float* { return (const float*)d_in[idx++]; };
  auto readDC = [&]() -> DCp {
    DCp d; d.w1=F(); d.b1=F(); d.g1=F(); d.be1=F(); d.w2=F(); d.b2=F(); d.g2=F(); d.be2=F();
    return d;
  };

  const float* input = F();
  DCp init0 = readDC();
  struct DownP { const float *w,*b; DCp dc; } p[4];
  for (int i = 0; i < 4; ++i){ p[i].w=F(); p[i].b=F(); p[i].dc=readDC(); }
  const float* m2_0=F(); const float* m2_1=F();
  const float* m1_0=F(); const float* m1_1=F(); const float* m1_2=F();
  const float* m0_0=F(); const float* m0_1=F(); const float* m0_2=F();
  const float* m_0 =F(); const float* m_1 =F(); const float* m_2 =F();
  struct UpP { const float *wt,*bt; DCp dc; } u[4];
  for (int i = 0; i < 4; ++i){ u[i].wt=F(); u[i].bt=F(); u[i].dc=readDC(); }
  const float* wf = F(); const float* bf = F();

  // ---- workspace arena ----
  char* base = (char*)d_ws; size_t off = 0;
  auto alloc = [&](size_t bytes) -> void* {
    size_t a = (off + 255) & ~(size_t)255; off = a + bytes; return (void*)(base + a);
  };
  float* a_init = (float*)alloc((size_t)2097152 * 4);  // [2,64,128,128]
  float* out3   = (float*)alloc((size_t)131072  * 4);  // [2,1024,8,8]
  float* catA   = (float*)alloc((size_t)262144  * 4);  // [2,2048,8,8]
  float* cat1   = (float*)alloc((size_t)1048576 * 4);  // [2,2048,16,16]
  float* cat2   = (float*)alloc((size_t)2097152 * 4);  // [2,1024,32,32]
  float* cat3   = (float*)alloc((size_t)4194304 * 4);  // [2,512,64,64]
  float* tmpA   = (float*)alloc((size_t)8388608 * 4);  // up to [2,256,128,128]
  float* tmpB   = (float*)alloc((size_t)2097152 * 4);  // up to [2,64,128,128]
  Scratch sc;
  sc.wbf   = (u16*)alloc((size_t)18874368 * 2);        // 1024 x 18432
  sc.col   = (u16*)alloc((size_t)COL_ELEMS * 2);
  sc.mbf   = (u16*)alloc((size_t)MEMP * 4096 * 2);
  sc.mt    = (u16*)alloc((size_t)MEMP * 4096 * 2);
  sc.qbf   = (u16*)alloc((size_t)2097152 * 2);
  sc.att   = (float*)alloc((size_t)1024 * MEMP * 4);
  sc.attbf = (u16*)alloc((size_t)1024 * MEMP * 2);
  sc.ybuf  = (float*)alloc((size_t)1048576 * 4);
  if (off > ws_size) return; // not enough scratch

  float* o    = (float*)d_out;
  float* XF   = o;                  // 2*3*128*128
  float* OUT0 = XF   + 98304;       // 2*128*64*64
  float* OUT1 = OUT0 + 1048576;     // 2*256*32*32
  float* OUT2 = OUT1 + 524288;      // 2*512*16*16
  float* X3   = OUT2 + 262144;      // 2*128*64*64
  float* X2   = X3   + 1048576;     // 2*256*32*32
  float* X1   = X2   + 524288;      // 2*512*16*16

  hipStream_t s = stream;

  // ---- encoder ----
  conv_layer(s, sc, input, 2, 3, 128, 128, init0.w1, init0.b1, init0.g1, init0.be1, 1, 1, 0, tmpB,   64, 0, 64);
  conv_layer(s, sc, tmpB,  2, 64,128, 128, init0.w2, init0.b2, init0.g2, init0.be2, 1, 1, 0, a_init, 64, 0, 64);

  conv_layer(s, sc, a_init, 2, 64, 128, 128, p[0].w, p[0].b, nullptr, nullptr, 0, 2, 0, tmpA, 128, 0, 128);
  conv_layer(s, sc, tmpA,   2, 128, 64, 64, p[0].dc.w1, p[0].dc.b1, p[0].dc.g1, p[0].dc.be1, 1, 1, 0, tmpB, 128, 0, 128);
  conv_layer(s, sc, tmpB,   2, 128, 64, 64, p[0].dc.w2, p[0].dc.b2, p[0].dc.g2, p[0].dc.be2, 1, 1, 0, OUT0, 128, 0, 128);

  conv_layer(s, sc, OUT0, 2, 128, 64, 64, p[1].w, p[1].b, nullptr, nullptr, 0, 2, 0, tmpA, 256, 0, 256);
  conv_layer(s, sc, tmpA, 2, 256, 32, 32, p[1].dc.w1, p[1].dc.b1, p[1].dc.g1, p[1].dc.be1, 1, 1, 0, tmpB, 256, 0, 256);
  conv_layer(s, sc, tmpB, 2, 256, 32, 32, p[1].dc.w2, p[1].dc.b2, p[1].dc.g2, p[1].dc.be2, 1, 1, 0, OUT1, 256, 0, 256);

  conv_layer(s, sc, OUT1, 2, 256, 32, 32, p[2].w, p[2].b, nullptr, nullptr, 0, 2, 0, tmpA, 512, 0, 512);
  conv_layer(s, sc, tmpA, 2, 512, 16, 16, p[2].dc.w1, p[2].dc.b1, p[2].dc.g1, p[2].dc.be1, 1, 1, 0, tmpB, 512, 0, 512);
  conv_layer(s, sc, tmpB, 2, 512, 16, 16, p[2].dc.w2, p[2].dc.b2, p[2].dc.g2, p[2].dc.be2, 1, 1, 0, OUT2, 512, 0, 512);

  conv_layer(s, sc, OUT2, 2, 512, 16, 16, p[3].w, p[3].b, nullptr, nullptr, 0, 2, 0, tmpA, 1024, 0, 1024);
  conv_layer(s, sc, tmpA, 2, 1024, 8, 8, p[3].dc.w1, p[3].dc.b1, p[3].dc.g1, p[3].dc.be1, 1, 1, 0, tmpB, 1024, 0, 1024);
  conv_layer(s, sc, tmpB, 2, 1024, 8, 8, p[3].dc.w2, p[3].dc.b2, p[3].dc.g2, p[3].dc.be2, 1, 1, 0, out3, 1024, 0, 1024);

  // ---- memory modules -> concat buffers ----
  mem_module(s, sc, out3, 2, 1024, 8, 8,  m2_0,  512,  2,  8, 1, catA, 2048, 0);
  mem_module(s, sc, out3, 2, 1024, 8, 8,  m2_1,  256,  1,  0, 0, catA, 2048, 1024);
  mem_module(s, sc, OUT2, 2, 512, 16, 16, m1_1, 2048,  8, 16, 1, cat1, 2048, 0);
  mem_module(s, sc, OUT2, 2, 512, 16, 16, m1_0, 1024,  4,  8, 1, cat1, 2048, 512);
  mem_module(s, sc, OUT2, 2, 512, 16, 16, m1_2, 1024,  1,  0, 0, cat1, 2048, 1024);
  mem_module(s, sc, OUT1, 2, 256, 32, 32, m0_1, 2048, 16, 32, 1, cat2, 1024, 0);
  mem_module(s, sc, OUT1, 2, 256, 32, 32, m0_0, 2048,  8,  8, 1, cat2, 1024, 256);
  mem_module(s, sc, OUT1, 2, 256, 32, 32, m0_2, 1024,  2,  0, 0, cat2, 1024, 512);
  mem_module(s, sc, OUT0, 2, 128, 64, 64, m_0,  2048, 16, 16, 1, cat3, 512, 0);
  mem_module(s, sc, OUT0, 2, 128, 64, 64, m_1,  4096, 32, 32, 1, cat3, 512, 128);
  mem_module(s, sc, OUT0, 2, 128, 64, 64, m_2,  1024,  4,  0, 0, cat3, 512, 256);

  // ---- decoder ----
  // u0: [2,2048,8,8] -> x1 [2,512,16,16] into cat1 @1536
  conv_layer(s, sc, catA, 2, 2048, 8, 8, u[0].wt, u[0].bt, nullptr, nullptr, 0, 1, 1, tmpA, 1024, 0, 1024);
  conv_layer(s, sc, tmpA, 2, 1024, 16, 16, u[0].dc.w1, u[0].dc.b1, u[0].dc.g1, u[0].dc.be1, 1, 1, 0, tmpB, 512, 0, 512);
  conv_layer(s, sc, tmpB, 2, 512, 16, 16, u[0].dc.w2, u[0].dc.b2, u[0].dc.g2, u[0].dc.be2, 1, 1, 0, cat1, 2048, 1536, 512);
  { long tot = (long)2 * 512 * 256;
    k_copych<<<cdiv(tot, 256), 256, 0, s>>>(cat1, 2048, 1536, 512, 256, X1, tot); }

  // u1: [2,2048,16,16] -> x2 [2,256,32,32] into cat2 @768
  conv_layer(s, sc, cat1, 2, 2048, 16, 16, u[1].wt, u[1].bt, nullptr, nullptr, 0, 1, 1, tmpA, 1024, 0, 1024);
  conv_layer(s, sc, tmpA, 2, 1024, 32, 32, u[1].dc.w1, u[1].dc.b1, u[1].dc.g1, u[1].dc.be1, 1, 1, 0, tmpB, 256, 0, 256);
  conv_layer(s, sc, tmpB, 2, 256, 32, 32, u[1].dc.w2, u[1].dc.b2, u[1].dc.g2, u[1].dc.be2, 1, 1, 0, cat2, 1024, 768, 256);
  { long tot = (long)2 * 256 * 1024;
    k_copych<<<cdiv(tot, 256), 256, 0, s>>>(cat2, 1024, 768, 256, 1024, X2, tot); }

  // u2: [2,1024,32,32] -> x3 [2,128,64,64] into cat3 @384
  conv_layer(s, sc, cat2, 2, 1024, 32, 32, u[2].wt, u[2].bt, nullptr, nullptr, 0, 1, 1, tmpA, 512, 0, 512);
  conv_layer(s, sc, tmpA, 2, 512, 64, 64, u[2].dc.w1, u[2].dc.b1, u[2].dc.g1, u[2].dc.be1, 1, 1, 0, tmpB, 128, 0, 128);
  conv_layer(s, sc, tmpB, 2, 128, 64, 64, u[2].dc.w2, u[2].dc.b2, u[2].dc.g2, u[2].dc.be2, 1, 1, 0, cat3, 512, 384, 128);
  { long tot = (long)2 * 128 * 4096;
    k_copych<<<cdiv(tot, 256), 256, 0, s>>>(cat3, 512, 384, 128, 4096, X3, tot); }

  // u3: [2,512,64,64] -> [2,64,128,128], then final 1x1 conv -> xf
  conv_layer(s, sc, cat3, 2, 512, 64, 64, u[3].wt, u[3].bt, nullptr, nullptr, 0, 1, 1, tmpA, 256, 0, 256);
  conv_layer(s, sc, tmpA, 2, 256, 128, 128, u[3].dc.w1, u[3].dc.b1, u[3].dc.g1, u[3].dc.be1, 1, 1, 0, tmpB, 64, 0, 64);
  conv_layer(s, sc, tmpB, 2, 64, 128, 128, u[3].dc.w2, u[3].dc.b2, u[3].dc.g2, u[3].dc.be2, 1, 1, 0, a_init, 64, 0, 64);
  k_conv1x1<<<cdiv((long)2 * 3 * 16384, 256), 256, 0, s>>>(a_init, wf, bf, XF);
}
